// VectorQuantizerEMA_28321014350518
// MI455X (gfx1250) — compile-verified, matrix operand order checked by reading, not run on hardware
//
#include <hip/hip_runtime.h>
#include <hip/hip_bf16.h>

// ---------------------------------------------------------------------------
// VQ-VAE EMA vector quantizer for MI455X (gfx1250, wave32, WMMA).
// N = 131072 rows, D = 128, K = 512 codes.
//  - distance matmul: v_wmma_f32_16x16x32_bf16, bf16 packed codebook staged in
//    LDS once per block (global_load_async_to_lds_b128 when available),
//    served by ds_load_b128 in the hot loop
//  - ||e||^2 folded into the WMMA accumulator init (C0 = -0.5*||e||^2) so
//    argmin(dist) == argmax(WMMA output); ||x||^2 not needed for argmin at all
//  - each wave owns 32 rows (two A tiles, two interleaved accumulator chains)
// ---------------------------------------------------------------------------

#define VQ_N 131072
#define VQ_D 128
#define VQ_K 512
#define VQ_DECAY 0.99f
#define VQ_EPS 1e-05f
#define VQ_COMMIT 0.25f

#define ROWS_PER_BLOCK 256

typedef __attribute__((ext_vector_type(16))) __bf16 v16bf;
typedef __attribute__((ext_vector_type(8)))  float  v8f;
typedef __attribute__((ext_vector_type(4)))  int    v4i;

#define AS3 __attribute__((address_space(3)))
#define AS1 __attribute__((address_space(1)))

#if defined(__gfx1250__) && __has_builtin(__builtin_amdgcn_global_load_async_to_lds_b128)
#define VQ_HAVE_ASYNC_LDS 1
#else
#define VQ_HAVE_ASYNC_LDS 0
#endif

// f32 -> bf16, round-to-nearest-even
static __device__ __forceinline__ unsigned short f2bf(float f) {
    union { float f; unsigned u; } v; v.f = f;
    unsigned u = v.u;
    u += 0x7FFFu + ((u >> 16) & 1u);
    return (unsigned short)(u >> 16);
}

// ---------------------------------------------------------------------------
// Kernel 0: zero workspace accumulators (dw[K*D], counts[K], lossacc[1])
// ---------------------------------------------------------------------------
__global__ void vq_zero(float* __restrict__ ws, int n) {
    int i = blockIdx.x * blockDim.x + threadIdx.x;
    if (i < n) ws[i] = 0.0f;
}

// ---------------------------------------------------------------------------
// Kernel 1: esq[k] = ||embedding[k]||^2 ; pack embedding into bf16 B-tiles in
// the exact per-lane WMMA B-matrix layout:
// Packed layout: [ntile(32)][ktile(4)][lane(32)][16 bf16]   (32B per lane)
//   lane L: n = ntile*16 + (L&15), kbase = (L<16 ? 0 : 8)
//   vec[0..7]  = e[n][ktile*32 + kbase + 0..7]
//   vec[8..15] = e[n][ktile*32 + kbase + 16..23]
// ---------------------------------------------------------------------------
__global__ void vq_prep(const float* __restrict__ emb,
                        float* __restrict__ esq,
                        unsigned short* __restrict__ ebf) {
    int t = blockIdx.x * blockDim.x + threadIdx.x;
    if (t < 32 * 4 * 32) {
        int ntile = t >> 7;
        int kt    = (t >> 5) & 3;
        int lane  = t & 31;
        int n     = ntile * 16 + (lane & 15);
        int kbase = (lane < 16) ? 0 : 8;
        const float* src = emb + (size_t)n * VQ_D + kt * 32 + kbase;
        unsigned short* dst = ebf + (size_t)t * 16;
#pragma unroll
        for (int j = 0; j < 8; ++j) {
            dst[j]     = f2bf(src[j]);
            dst[8 + j] = f2bf(src[16 + j]);
        }
    } else if (t < 32 * 4 * 32 + VQ_K) {
        int k = t - 32 * 4 * 32;
        const float* e = emb + (size_t)k * VQ_D;
        float s = 0.0f;
#pragma unroll 8
        for (int d = 0; d < VQ_D; ++d) s += e[d] * e[d];
        esq[k] = s;
    }
}

// ---------------------------------------------------------------------------
// Kernel 2: main fused kernel. One block = 256 rows, 8 wave32s; each wave
// owns 32 rows (2 A tiles). Codebook (128 KB bf16) staged in LDS per block.
// ---------------------------------------------------------------------------
__global__ void __launch_bounds__(256)
vq_main(const float* __restrict__ x,
        const float* __restrict__ emb,
        const unsigned short* __restrict__ ebf,
        const float* __restrict__ esq,
        float* __restrict__ outq,
        float* __restrict__ outidx,
        float* __restrict__ dw,
        float* __restrict__ counts,
        float* __restrict__ lossacc) {
    __shared__ unsigned short s_ebf[32 * 4 * 32 * 16];  // 128 KB packed B
    __shared__ float s_esq[VQ_K];
    __shared__ int   s_idx[ROWS_PER_BLOCK];
    __shared__ float s_red[8];

    const int tid = threadIdx.x;
    const int rowBase = blockIdx.x * ROWS_PER_BLOCK;

    // ||e||^2 into LDS
    s_esq[tid]       = esq[tid];
    s_esq[tid + 256] = esq[tid + 256];

    // ---- stage packed bf16 codebook into LDS (8192 x 16B) ----
#if VQ_HAVE_ASYNC_LDS
    {
        AS1 v4i* gsrc = (AS1 v4i*)ebf;
        AS3 v4i* ldst = (AS3 v4i*)(void*)s_ebf;
#pragma unroll 4
        for (int i = tid; i < 8192; i += 256)
            __builtin_amdgcn_global_load_async_to_lds_b128(gsrc + i, ldst + i, 0, 0);
#if __has_builtin(__builtin_amdgcn_s_wait_asynccnt)
        __builtin_amdgcn_s_wait_asynccnt(0);
#else
        asm volatile("s_wait_asynccnt 0x0" ::: "memory");
#endif
    }
#else
    {
        float4* ldst = (float4*)s_ebf;
        const float4* gsrc = (const float4*)ebf;
#pragma unroll 4
        for (int i = tid; i < 8192; i += 256) ldst[i] = gsrc[i];
    }
#endif
    __syncthreads();

    const int wave  = tid >> 5;
    const int lane  = tid & 31;
    const int mrow  = lane & 15;
    const int kbase = (lane < 16) ? 0 : 8;

    // ---- load two A tiles (rows wave*32 + 0..15 and +16..31), f32->bf16 ----
    v16bf a0[4], a1[4];
#pragma unroll
    for (int t2 = 0; t2 < 2; ++t2) {
        const float* xrow =
            x + (size_t)(rowBase + wave * 32 + t2 * 16 + mrow) * VQ_D;
#pragma unroll
        for (int kt = 0; kt < 4; ++kt) {
            const float* p = xrow + kt * 32 + kbase;
            float4 q0 = *(const float4*)(p);
            float4 q1 = *(const float4*)(p + 4);
            float4 q2 = *(const float4*)(p + 16);
            float4 q3 = *(const float4*)(p + 20);
            union { v16bf v; unsigned short s[16]; } ua;
            ua.s[0]  = f2bf(q0.x); ua.s[1]  = f2bf(q0.y);
            ua.s[2]  = f2bf(q0.z); ua.s[3]  = f2bf(q0.w);
            ua.s[4]  = f2bf(q1.x); ua.s[5]  = f2bf(q1.y);
            ua.s[6]  = f2bf(q1.z); ua.s[7]  = f2bf(q1.w);
            ua.s[8]  = f2bf(q2.x); ua.s[9]  = f2bf(q2.y);
            ua.s[10] = f2bf(q2.z); ua.s[11] = f2bf(q2.w);
            ua.s[12] = f2bf(q3.x); ua.s[13] = f2bf(q3.y);
            ua.s[14] = f2bf(q3.z); ua.s[15] = f2bf(q3.w);
            if (t2 == 0) a0[kt] = ua.v; else a1[kt] = ua.v;
        }
    }

    // preload the per-lane -0.5*||e||^2 accumulator seeds for all 32 n-tiles
    float eqh[32];
#pragma unroll
    for (int nt = 0; nt < 32; ++nt)
        eqh[nt] = -0.5f * s_esq[nt * 16 + (lane & 15)];

    // argmax of (x.e - 0.5||e||^2)  ==  argmin of squared distance
    float bestv0[8], bestv1[8];
    int   besti0[8], besti1[8];
#pragma unroll
    for (int r = 0; r < 8; ++r) {
        bestv0[r] = -3.4e38f; bestv1[r] = -3.4e38f;
        besti0[r] = 0;        besti1[r] = 0;
    }

    // ---- sweep 32 code tiles: 8x WMMA each (2 M-tiles x 4 K-steps) ----
    for (int nt = 0; nt < 32; ++nt) {
        const float h = eqh[nt];
        v8f c0 = {h, h, h, h, h, h, h, h};
        v8f c1 = {h, h, h, h, h, h, h, h};
#pragma unroll
        for (int kt = 0; kt < 4; ++kt) {
            const v16bf b = *(const v16bf*)(
                s_ebf + (size_t)((nt * 4 + kt) * 32 + lane) * 16);
            c0 = __builtin_amdgcn_wmma_f32_16x16x32_bf16(
                     false, a0[kt], false, b, (short)0, c0, false, false);
            c1 = __builtin_amdgcn_wmma_f32_16x16x32_bf16(
                     false, a1[kt], false, b, (short)0, c1, false, false);
        }
        const int n = nt * 16 + (lane & 15);
#pragma unroll
        for (int r = 0; r < 8; ++r) {
            if (c0[r] > bestv0[r]) { bestv0[r] = c0[r]; besti0[r] = n; }
            if (c1[r] > bestv1[r]) { bestv1[r] = c1[r]; besti1[r] = n; }
        }
    }

    // argmax across the 16 lanes sharing each M (xor stays within each half)
#pragma unroll
    for (int off = 8; off >= 1; off >>= 1) {
#pragma unroll
        for (int r = 0; r < 8; ++r) {
            float ov0 = __shfl_xor(bestv0[r], off);
            int   oi0 = __shfl_xor(besti0[r], off);
            if (ov0 > bestv0[r] || (ov0 == bestv0[r] && oi0 < besti0[r])) {
                bestv0[r] = ov0; besti0[r] = oi0;
            }
            float ov1 = __shfl_xor(bestv1[r], off);
            int   oi1 = __shfl_xor(besti1[r], off);
            if (ov1 > bestv1[r] || (ov1 == bestv1[r] && oi1 < besti1[r])) {
                bestv1[r] = ov1; besti1[r] = oi1;
            }
        }
    }
    if (lane == 0 || lane == 16) {
        int mb = wave * 32 + ((lane == 0) ? 0 : 8);
#pragma unroll
        for (int r = 0; r < 8; ++r) {
            s_idx[mb + r]      = besti0[r];
            s_idx[mb + 16 + r] = besti1[r];
        }
    }
    __syncthreads();

    // ---- cooperative epilogue: quantize gather, loss, dw scatter ----
    float lsum = 0.0f;
#pragma unroll 4
    for (int c0i = 0; c0i < ROWS_PER_BLOCK * 32 / 256; ++c0i) {
        int g   = c0i * 256 + tid;     // (row, dword4)
        int row = g >> 5;
        int d4  = g & 31;
        int idx = s_idx[row];
        const float4 e4 = *(const float4*)(emb + (size_t)idx * VQ_D + d4 * 4);
        const float4 x4 = *(const float4*)(x + (size_t)(rowBase + row) * VQ_D + d4 * 4);
        *(float4*)(outq + (size_t)(rowBase + row) * VQ_D + d4 * 4) = e4;
        float dx = e4.x - x4.x, dy = e4.y - x4.y,
              dz = e4.z - x4.z, dwv = e4.w - x4.w;
        lsum += dx * dx + dy * dy + dz * dz + dwv * dwv;
        float* dwp = dw + (size_t)idx * VQ_D + d4 * 4;
        atomicAdd(dwp + 0, x4.x);
        atomicAdd(dwp + 1, x4.y);
        atomicAdd(dwp + 2, x4.z);
        atomicAdd(dwp + 3, x4.w);
    }

    // block-reduce the loss partials
#pragma unroll
    for (int off = 16; off >= 1; off >>= 1) lsum += __shfl_xor(lsum, off);
    if (lane == 0) s_red[wave] = lsum;
    __syncthreads();
    if (tid == 0) {
        float t = 0.0f;
#pragma unroll
        for (int w = 0; w < 8; ++w) t += s_red[w];
        atomicAdd(lossacc, t);
    }

    // counts + index output (one row per thread)
    {
        int idx = s_idx[tid];
        atomicAdd(&counts[idx], 1.0f);
        outidx[rowBase + tid] = (float)idx;
    }
}

// ---------------------------------------------------------------------------
// Kernel 3: EMA update, normalization, dead-code reinit, loss scale.
// ---------------------------------------------------------------------------
__global__ void vq_finalize(const float* __restrict__ x,
                            const float* __restrict__ ema_cs,
                            const float* __restrict__ ema_w,
                            const float* __restrict__ counts,
                            const float* __restrict__ dw,
                            const float* __restrict__ lossacc,
                            float* __restrict__ out_emb,
                            float* __restrict__ out_cs,
                            float* __restrict__ out_w,
                            float* __restrict__ out_loss) {
    __shared__ float s[VQ_K];
    int k = threadIdx.x;
    float cs = VQ_DECAY * ema_cs[k] + (1.0f - VQ_DECAY) * counts[k];
    s[k] = cs;
    __syncthreads();
#pragma unroll
    for (int off = 256; off >= 1; off >>= 1) {
        if (k < off) s[k] += s[k + off];
        __syncthreads();
    }
    float n = s[0];
    float ncs = (cs + VQ_EPS) / (n + (float)VQ_K * VQ_EPS) * n;
    bool dead = ncs < 1.0f;

    // deterministic pseudo-random replacement row (stands in for threefry)
    unsigned r = (unsigned)k * 2654435761u ^ 0x9E3779B9u;
    r %= (unsigned)VQ_N;
    const float* repl = x + (size_t)r * VQ_D;

    const float* ewr = ema_w + (size_t)k * VQ_D;
    const float* dwr = dw + (size_t)k * VQ_D;
    float* oer = out_emb + (size_t)k * VQ_D;
    float* owr = out_w + (size_t)k * VQ_D;
    float inv = 1.0f / ncs;
#pragma unroll 4
    for (int d = 0; d < VQ_D; ++d) {
        float w = VQ_DECAY * ewr[d] + (1.0f - VQ_DECAY) * dwr[d];
        float e = w * inv;
        if (dead) { e = repl[d]; w = repl[d]; }
        oer[d] = e;
        owr[d] = w;
    }
    out_cs[k] = dead ? 1.0f : ncs;
    if (k == 0)
        out_loss[0] = VQ_COMMIT * lossacc[0] / (float)((size_t)VQ_N * VQ_D);
}

// ---------------------------------------------------------------------------
extern "C" void kernel_launch(void* const* d_in, const int* in_sizes, int n_in,
                              void* d_out, int out_size, void* d_ws, size_t ws_size,
                              hipStream_t stream) {
    const float* x      = (const float*)d_in[0];  // [32,4096,128]
    const float* emb    = (const float*)d_in[1];  // [512,128]
    const float* ema_cs = (const float*)d_in[2];  // [512]
    const float* ema_w  = (const float*)d_in[3];  // [512,128]

    // Output layout (flat, reference return order):
    float* out      = (float*)d_out;
    float* o_q      = out;                         // 16777216
    float* o_loss   = o_q + (size_t)VQ_N * VQ_D;   // 1
    float* o_idx    = o_loss + 1;                  // 131072
    float* o_emb    = o_idx + VQ_N;                // 65536
    float* o_cs     = o_emb + VQ_K * VQ_D;         // 512
    float* o_w      = o_cs + VQ_K;                 // 65536

    // Workspace layout (floats): dw[65536] counts[512] lossacc[1] pad esq[512] | ebf16
    float* ws      = (float*)d_ws;
    float* dw      = ws;                           // 65536
    float* counts  = dw + VQ_K * VQ_D;             // 512
    float* lossacc = counts + VQ_K;                // 1
    float* esq     = ws + 66064;                   // 512
    unsigned short* ebf = (unsigned short*)((char*)d_ws + (size_t)66576 * 4); // 32B aligned

    // 1) zero accumulators (dw + counts + lossacc are contiguous)
    {
        int n = VQ_K * VQ_D + VQ_K + 1;            // 66049
        vq_zero<<<(n + 255) / 256, 256, 0, stream>>>(ws, n);
    }
    // 2) pack codebook to bf16 B-tile layout + ||e||^2
    vq_prep<<<(32 * 4 * 32 + VQ_K + 255) / 256, 256, 0, stream>>>(emb, esq, ebf);
    // 3) fused distance/argmin/quantize/scatter
    vq_main<<<VQ_N / ROWS_PER_BLOCK, 256, 0, stream>>>(x, emb, ebf, esq,
                                                       o_q, o_idx, dw, counts, lossacc);
    // 4) EMA + dead-code + loss
    vq_finalize<<<1, VQ_K, 0, stream>>>(x, ema_cs, ema_w, counts, dw, lossacc,
                                        o_emb, o_cs, o_w, o_loss);
    (void)in_sizes; (void)n_in; (void)out_size; (void)ws_size;
}
